// ExpertPool_90632399880280
// MI455X (gfx1250) — matrix operand order repaired and checked by reading, not compiled, not run on hardware
//
#include <hip/hip_runtime.h>
#include <math.h>

typedef __attribute__((ext_vector_type(16))) __bf16 v16bf;
typedef __attribute__((ext_vector_type(8)))  __bf16 v8bf;
typedef __attribute__((ext_vector_type(4)))  __bf16 v4bf;
typedef __attribute__((ext_vector_type(8)))  float  v8f;
typedef __attribute__((ext_vector_type(4)))  unsigned int v4ui;
typedef __attribute__((ext_vector_type(8)))  int v8i;
typedef __attribute__((ext_vector_type(4)))  int v4i;

#define T_DIM 4096
#define H_DIM 1024
#define F_DIM 4096
#define NEXP  8
#define TOPK  2

__device__ __forceinline__ float gelu_exact(float x) {
    return 0.5f * x * (1.0f + erff(x * 0.70710678118654752440f));
}

// ---------------------------------------------------------------- combine wts
__global__ void moe_cw_kernel(const float* __restrict__ ew,
                              const int* __restrict__ sel,
                              float* __restrict__ cw) {
    int i = blockIdx.x * blockDim.x + threadIdx.x;   // over T*NEXP
    if (i >= T_DIM * NEXP) return;
    int t = i / NEXP;
    int e = i - t * NEXP;
    float s = 0.0f;
#pragma unroll
    for (int k = 0; k < TOPK; ++k) {
        int se = sel[t * TOPK + k];
        if (se == e && se >= 0) s += ew[t * TOPK + k];
    }
    cw[i] = s;
}

// ---------------------------------------------------------------- fp32->bf16
__global__ void convert_bf16_kernel(const float* __restrict__ in,
                                    __bf16* __restrict__ out, int n8) {
    int i = blockIdx.x * blockDim.x + threadIdx.x;
    if (i >= n8) return;
    const float* p = in + (size_t)i * 8;
    float4 f0 = *(const float4*)p;
    float4 f1 = *(const float4*)(p + 4);
    v8bf o;
    o[0] = (__bf16)f0.x; o[1] = (__bf16)f0.y; o[2] = (__bf16)f0.z; o[3] = (__bf16)f0.w;
    o[4] = (__bf16)f1.x; o[5] = (__bf16)f1.y; o[6] = (__bf16)f1.z; o[7] = (__bf16)f1.w;
    *(v8bf*)(out + (size_t)i * 8) = o;
}

// -------------------------------------------- batched transpose + fp32->bf16
// in:  [z][R][C] fp32  ->  out: [z][C][R] bf16   (coalesced both sides via LDS)
#define TT 64
__global__ void transpose_convert_kernel(const float* __restrict__ in,
                                         __bf16* __restrict__ out,
                                         int R, int C) {
    __shared__ float tile[TT][TT + 1];
    const float* inp = in + (size_t)blockIdx.z * R * C;
    __bf16* op = out + (size_t)blockIdx.z * R * C;
    const int r0 = blockIdx.x * TT;
    const int c0 = blockIdx.y * TT;
    const int cx = threadIdx.x & 15;   // 0..15
    const int cy = threadIdx.x >> 4;   // 0..15
#pragma unroll
    for (int j = 0; j < 4; ++j) {
        int r = cy + 16 * j;
        float4 v = *(const float4*)(inp + (size_t)(r0 + r) * C + c0 + cx * 4);
        tile[r][cx * 4 + 0] = v.x;
        tile[r][cx * 4 + 1] = v.y;
        tile[r][cx * 4 + 2] = v.z;
        tile[r][cx * 4 + 3] = v.w;
    }
    __syncthreads();
#pragma unroll
    for (int j = 0; j < 4; ++j) {
        int c = cy + 16 * j;           // output row = input column
        v4bf vv;
#pragma unroll
        for (int q = 0; q < 4; ++q) vv[q] = (__bf16)tile[cx * 4 + q][c];
        *(v4bf*)(op + (size_t)(c0 + c) * R + r0 + cx * 4) = vv;
    }
}

// -------------------------------------------------------- TDM 2D tile -> LDS
// D# per CDNA5 ISA §8: group0 = {count/flags, lds_addr, global_addr, type=2},
// group1 = {data_size=2B, pad(16DW interval, 4DW amount), dims, tile, stride}.
// Loads a tile_d1 x tile_d0 (rows x elems) bf16 tile whose rows are
// stride_d0 elements apart, into LDS with 64B data + 16B pad per row
// (i.e. LDSS=40 bf16 row stride).
__device__ __forceinline__ void tdm_load_tile_2d(unsigned lds_addr,
                                                 const __bf16* gptr,
                                                 unsigned tensor_d0,
                                                 unsigned tensor_d1,
                                                 unsigned tile_d0,
                                                 unsigned tile_d1,
                                                 unsigned long long stride_d0) {
    unsigned long long ga = (unsigned long long)(const void*)gptr;
    v4ui g0;
    g0[0] = 1u;                                            // count=1 (valid user D#)
    g0[1] = lds_addr;                                      // LDS byte address
    g0[2] = (unsigned)ga;                                  // global addr [31:0]
    g0[3] = (unsigned)((ga >> 32) & 0x01ffffffu)           // global addr [56:32]
          | (2u << 30);                                    // type = 2 ("image")
    const unsigned w0 = (1u << 16)    // data_size: 2 bytes
                      | (1u << 20)    // pad_enable
                      | (3u << 22)    // pad_interval: 16 DWORDs (64B) per pad
                      | (3u << 25);   // pad_amount:   4 DWORDs (16B)
    v8i g1;
    g1[0] = (int)w0;
    g1[1] = (int)((tensor_d0 & 0xffffu) << 16);            // tensor_dim0[15:0] @63:48
    g1[2] = (int)((tensor_d0 >> 16) | ((tensor_d1 & 0xffffu) << 16));
    g1[3] = (int)((tensor_d1 >> 16) | ((tile_d0 & 0xffffu) << 16));
    g1[4] = (int)(tile_d1 & 0xffffu);                      // tile_dim1; tile_dim2=0
    g1[5] = (int)(unsigned)(stride_d0 & 0xffffffffu);      // dim0_stride[31:0]
    g1[6] = (int)(unsigned)((stride_d0 >> 32) & 0xffffu);  // dim0_stride[47:32]
    g1[7] = 0;                                             // dim1_stride = 0
    v4i zz4 = {};
    v8i zz8 = {};
    __builtin_amdgcn_tensor_load_to_lds(g0, g1, zz4, zz4, zz8, 0);
}

// ============================================================================
// FAST PATH: bf16 GEMM. A-tile staged via per-lane async-to-LDS copies,
// B-tile staged via a single Tensor Data Mover descriptor (wave 0).
//   A  : [M x Kd] bf16 row-major
//   Bt : [N x Kd] bf16 row-major (pre-transposed weights)
// MODE 0: Cbf = bf16(gelu(acc + bias))
// MODE 1: Cf (+)= cw[row,e] * (acc + bias)
// ============================================================================
#define GBM 128
#define GBN 256
#define GBK 32
#define LDSS 40          // padded LDS row stride in bf16 elems (80B, 16B mult)
#define GTHREADS 512     // 16 waves: 2 x 8, each wave owns 64x32

template <int MODE>
__launch_bounds__(GTHREADS)
__global__ void moe_gemm_tdm(const __bf16* __restrict__ A,
                             const __bf16* __restrict__ Bt,
                             const float* __restrict__ bias,
                             __bf16* __restrict__ Cbf,
                             float* __restrict__ Cf,
                             const float* __restrict__ cw,
                             int Kd, int N, int e, int accumulate) {
    __shared__ __align__(16) __bf16 As[2][GBM * LDSS];
    __shared__ __align__(16) __bf16 Bs[2][GBN * LDSS];

    const int tid  = threadIdx.x;
    const int wave = tid >> 5;
    const int lane = tid & 31;
    const int wm   = wave >> 3;      // 0..1
    const int wn   = wave & 7;       // 0..7
    const int half = lane >> 4;
    const int l16  = lane & 15;

    const int rowBase = blockIdx.x * GBM;
    const int colBase = blockIdx.y * GBN;

    // A staging: 512 16B chunks, exactly one per thread
    const int ar = tid >> 2;          // 0..127
    const int ac = (tid & 3) * 8;     // 0,8,16,24

    auto issueA = [&](int kb, int buf) {
        unsigned lds = (unsigned)(unsigned long long)&As[buf][ar * LDSS + ac];
        const __bf16* gp = A + (size_t)(rowBase + ar) * Kd + kb + ac;
        asm volatile("global_load_async_to_lds_b128 %0, %1, off"
                     :: "v"(lds), "v"(gp) : "memory");
    };
    auto issueB = [&](int kb, int buf) {
        unsigned lds = (unsigned)(unsigned long long)&Bs[buf][0];
        tdm_load_tile_2d(lds, Bt + (size_t)colBase * Kd + kb,
                         (unsigned)Kd, (unsigned)N,
                         (unsigned)GBK, (unsigned)GBN,
                         (unsigned long long)Kd);
    };

    v8f acc[4][2] = {};
    const int nk = Kd / GBK;

    issueA(0, 0);
    if (wave == 0) issueB(0, 0);

    for (int i = 0; i < nk; ++i) {
        const int cur = i & 1;
        if (i + 1 < nk) {
            issueA((i + 1) * GBK, cur ^ 1);
            if (wave == 0) issueB((i + 1) * GBK, cur ^ 1);
            // newest op (next tile) may stay outstanding; oldest (this tile)
            // must have landed — async/TDM ops complete in order per wave.
            asm volatile("s_wait_asynccnt 0x1" ::: "memory");
            __builtin_amdgcn_s_wait_tensorcnt(1);
        } else {
            asm volatile("s_wait_asynccnt 0x0" ::: "memory");
            __builtin_amdgcn_s_wait_tensorcnt(0);
        }
        __syncthreads();

        v16bf afrag[4], bfrag[2];
#pragma unroll
        for (int m = 0; m < 4; ++m) {
            const __bf16* p = &As[cur][(wm * 64 + m * 16 + l16) * LDSS];
            ((v8bf*)&afrag[m])[0] = *(const v8bf*)(p + half * 8);       // K h*8..+7
            ((v8bf*)&afrag[m])[1] = *(const v8bf*)(p + 16 + half * 8);  // K 16+h*8..
        }
#pragma unroll
        for (int n = 0; n < 2; ++n) {
            const __bf16* p = &Bs[cur][(wn * 32 + n * 16 + l16) * LDSS];
            ((v8bf*)&bfrag[n])[0] = *(const v8bf*)(p + half * 8);       // K h*8..+7
            ((v8bf*)&bfrag[n])[1] = *(const v8bf*)(p + 16 + half * 8);  // K 16+h*8..
        }
#pragma unroll
        for (int m = 0; m < 4; ++m)
#pragma unroll
            for (int n = 0; n < 2; ++n)
                acc[m][n] = __builtin_amdgcn_wmma_f32_16x16x32_bf16(
                    false, afrag[m], false, bfrag[n],
                    (short)0, acc[m][n], false, false);
        __syncthreads();
    }

#pragma unroll
    for (int m = 0; m < 4; ++m) {
#pragma unroll
        for (int n = 0; n < 2; ++n) {
            const int col = colBase + wn * 32 + n * 16 + l16;
            const float bv = bias[col];
#pragma unroll
            for (int r = 0; r < 8; ++r) {
                const int row = rowBase + wm * 64 + m * 16 + r + 8 * half;
                float v = acc[m][n][r] + bv;
                if constexpr (MODE == 0) {
                    Cbf[(size_t)row * N + col] = (__bf16)gelu_exact(v);
                } else {
                    const float s = cw[row * NEXP + e];
                    const size_t idx = (size_t)row * N + col;
                    const float val = s * v;
                    if (accumulate) Cf[idx] += val;
                    else            Cf[idx]  = val;
                }
            }
        }
    }
}

// ============================================================================
// FALLBACK (small workspace): on-the-fly fp32->bf16 GEMM
// ============================================================================
#define BM 128
#define BN 128
#define BK 32
#define NTHREADS 512

template <typename SrcT, int MODE>
__launch_bounds__(NTHREADS)
__global__ void moe_gemm_wmma(const SrcT* __restrict__ A,
                              const float* __restrict__ B,
                              const float* __restrict__ bias,
                              __bf16* __restrict__ Cbf,
                              float* __restrict__ Cf,
                              const float* __restrict__ cw,
                              int Kd, int N, int e, int accumulate) {
    __shared__ __align__(16) __bf16 As[BM * LDSS];
    __shared__ __align__(16) __bf16 Bs[BN * LDSS];   // transposed: [n][k]

    const int tid  = threadIdx.x;
    const int wave = tid >> 5;
    const int lane = tid & 31;
    const int wm   = wave >> 2;
    const int wn   = wave & 3;
    const int half = lane >> 4;
    const int l16  = lane & 15;

    const int rowBase = blockIdx.x * BM;
    const int colBase = blockIdx.y * BN;

    v8f acc[2][2] = {};

    const int arow = tid >> 2;
    const int acol = (tid & 3) * 8;
    const int bk   = tid >> 4;
    const int bn   = (tid & 15) * 8;

    for (int kb = 0; kb < Kd; kb += BK) {
        {
            const SrcT* ap = A + (size_t)(rowBase + arow) * Kd + kb + acol;
            __bf16* dst = &As[arow * LDSS + acol];
            if constexpr (sizeof(SrcT) == 4) {
                float4 f0 = *(const float4*)ap;
                float4 f1 = *(const float4*)(ap + 4);
                dst[0] = (__bf16)f0.x; dst[1] = (__bf16)f0.y;
                dst[2] = (__bf16)f0.z; dst[3] = (__bf16)f0.w;
                dst[4] = (__bf16)f1.x; dst[5] = (__bf16)f1.y;
                dst[6] = (__bf16)f1.z; dst[7] = (__bf16)f1.w;
            } else {
                *(v8bf*)dst = *(const v8bf*)ap;
            }
        }
        {
            const float* bp = B + (size_t)(kb + bk) * N + colBase + bn;
            float4 f0 = *(const float4*)bp;
            float4 f1 = *(const float4*)(bp + 4);
            __bf16 v[8] = {(__bf16)f0.x, (__bf16)f0.y, (__bf16)f0.z, (__bf16)f0.w,
                           (__bf16)f1.x, (__bf16)f1.y, (__bf16)f1.z, (__bf16)f1.w};
#pragma unroll
            for (int j = 0; j < 8; ++j)
                Bs[(bn + j) * LDSS + bk] = v[j];
        }
        if (kb + BK < Kd) {
            __builtin_prefetch(A + (size_t)(rowBase + arow) * Kd + (kb + BK) + acol, 0, 1);
            __builtin_prefetch(B + (size_t)(kb + BK + bk) * N + colBase + bn, 0, 1);
        }
        __syncthreads();

        v16bf afrag[2], bfrag[2];
#pragma unroll
        for (int m = 0; m < 2; ++m) {
            const __bf16* p = &As[(wm * 32 + m * 16 + l16) * LDSS];
            ((v8bf*)&afrag[m])[0] = *(const v8bf*)(p + half * 8);
            ((v8bf*)&afrag[m])[1] = *(const v8bf*)(p + 16 + half * 8);
        }
#pragma unroll
        for (int n = 0; n < 2; ++n) {
            const __bf16* p = &Bs[(wn * 32 + n * 16 + l16) * LDSS];
            ((v8bf*)&bfrag[n])[0] = *(const v8bf*)(p + half * 8);
            ((v8bf*)&bfrag[n])[1] = *(const v8bf*)(p + 16 + half * 8);
        }
#pragma unroll
        for (int m = 0; m < 2; ++m)
#pragma unroll
            for (int n = 0; n < 2; ++n)
                acc[m][n] = __builtin_amdgcn_wmma_f32_16x16x32_bf16(
                    false, afrag[m], false, bfrag[n],
                    (short)0, acc[m][n], false, false);
        __syncthreads();
    }

#pragma unroll
    for (int m = 0; m < 2; ++m) {
#pragma unroll
        for (int n = 0; n < 2; ++n) {
            const int col = colBase + wn * 32 + n * 16 + l16;
            const float bv = bias[col];
#pragma unroll
            for (int r = 0; r < 8; ++r) {
                const int row = rowBase + wm * 32 + m * 16 + r + 8 * half;
                float v = acc[m][n][r] + bv;
                if constexpr (MODE == 0) {
                    Cbf[(size_t)row * N + col] = (__bf16)gelu_exact(v);
                } else {
                    const float s = cw[row * NEXP + e];
                    const size_t idx = (size_t)row * N + col;
                    const float val = s * v;
                    if (accumulate) Cf[idx] += val;
                    else            Cf[idx]  = val;
                }
            }
        }
    }
}

// ============================================================================
extern "C" void kernel_launch(void* const* d_in, const int* in_sizes, int n_in,
                              void* d_out, int out_size, void* d_ws, size_t ws_size,
                              hipStream_t stream) {
    const float* hs  = (const float*)d_in[0];   // [T,H]
    const float* ew  = (const float*)d_in[1];   // [T,K]
    const float* W1  = (const float*)d_in[2];   // [E,H,F]
    const float* b1  = (const float*)d_in[3];   // [E,F]
    const float* W2  = (const float*)d_in[4];   // [E,F,H]
    const float* b2  = (const float*)d_in[5];   // [E,H]
    const int*   sel = (const int*)d_in[6];     // [T,K]
    float* out = (float*)d_out;                 // [T,H]

    const size_t cw_b = (size_t)T_DIM * NEXP * 4;
    const size_t xb_b = (size_t)T_DIM * H_DIM * 2;
    const size_t h1_b = (size_t)T_DIM * F_DIM * 2;
    const size_t w1_b = (size_t)NEXP * H_DIM * F_DIM * 2;
    const size_t w2_b = (size_t)NEXP * F_DIM * H_DIM * 2;

    size_t off = 0;
    auto carve = [&](size_t bytes) {
        void* p = (char*)d_ws + off;
        off = (off + bytes + 255) & ~(size_t)255;
        return p;
    };

    float* cw = (float*)carve(cw_b);
    const size_t need_fast = ((cw_b + 255) & ~(size_t)255) +
                             ((xb_b + 255) & ~(size_t)255) +
                             ((h1_b + 255) & ~(size_t)255) +
                             ((w1_b + 255) & ~(size_t)255) +
                             ((w2_b + 255) & ~(size_t)255);

    moe_cw_kernel<<<(T_DIM * NEXP + 255) / 256, 256, 0, stream>>>(ew, sel, cw);

    if (ws_size >= need_fast) {
        __bf16* Xbf = (__bf16*)carve(xb_b);
        __bf16* H1  = (__bf16*)carve(h1_b);
        __bf16* W1t = (__bf16*)carve(w1_b);   // [E][F][H]
        __bf16* W2t = (__bf16*)carve(w2_b);   // [E][H][F]

        convert_bf16_kernel<<<(T_DIM * H_DIM / 8 + 255) / 256, 256, 0, stream>>>(
            hs, Xbf, T_DIM * H_DIM / 8);
        dim3 tg1(H_DIM / TT, F_DIM / TT, NEXP);
        transpose_convert_kernel<<<tg1, 256, 0, stream>>>(W1, W1t, H_DIM, F_DIM);
        dim3 tg2(F_DIM / TT, H_DIM / TT, NEXP);
        transpose_convert_kernel<<<tg2, 256, 0, stream>>>(W2, W2t, F_DIM, H_DIM);

        dim3 g1(T_DIM / GBM, F_DIM / GBN);   // 32 x 16
        dim3 g2(T_DIM / GBM, H_DIM / GBN);   // 32 x 4
        for (int e = 0; e < NEXP; ++e) {
            moe_gemm_tdm<0><<<g1, GTHREADS, 0, stream>>>(
                Xbf, W1t + (size_t)e * H_DIM * F_DIM, b1 + (size_t)e * F_DIM,
                H1, (float*)nullptr, (const float*)nullptr,
                H_DIM, F_DIM, e, 0);
            moe_gemm_tdm<1><<<g2, GTHREADS, 0, stream>>>(
                H1, W2t + (size_t)e * F_DIM * H_DIM, b2 + (size_t)e * H_DIM,
                (__bf16*)nullptr, out, cw,
                F_DIM, H_DIM, e, (e > 0) ? 1 : 0);
        }
    } else {
        __bf16* H1 = (__bf16*)carve(h1_b);
        dim3 g1(T_DIM / BM, F_DIM / BN);     // 32 x 32
        dim3 g2(T_DIM / BM, H_DIM / BN);     // 32 x 8
        for (int e = 0; e < NEXP; ++e) {
            moe_gemm_wmma<float, 0><<<g1, NTHREADS, 0, stream>>>(
                hs, W1 + (size_t)e * H_DIM * F_DIM, b1 + (size_t)e * F_DIM,
                H1, (float*)nullptr, (const float*)nullptr,
                H_DIM, F_DIM, e, 0);
            moe_gemm_wmma<__bf16, 1><<<g2, NTHREADS, 0, stream>>>(
                H1, W2 + (size_t)e * F_DIM * H_DIM, b2 + (size_t)e * H_DIM,
                (__bf16*)nullptr, out, cw,
                F_DIM, H_DIM, e, (e > 0) ? 1 : 0);
        }
    }
}
